// SEGNN_64725157151340
// MI455X (gfx1250) — matrix operand order repaired and checked by reading, not compiled, not run on hardware
//
#include <hip/hip_runtime.h>

typedef __bf16 bf16;
typedef __attribute__((ext_vector_type(16))) __bf16 bf16x16;
typedef __attribute__((ext_vector_type(8)))  __bf16 bf16x8;
typedef __attribute__((ext_vector_type(8)))  float  f32x8;

#define NNODES 20000
#define NEDGES 200000
#define NGRAPH 64

__device__ __forceinline__ float sigm(float x) { return 1.f / (1.f + __expf(-x)); }
__device__ __forceinline__ f32x8 fzero() { return (f32x8){0.f,0.f,0.f,0.f,0.f,0.f,0.f,0.f}; }

__device__ __forceinline__ f32x8 wmma_bf16(bf16x16 a, bf16x16 b, f32x8 c) {
  return __builtin_amdgcn_wmma_f32_16x16x32_bf16(false, a, false, b, (short)0, c,
                                                 false, false);
}

// ---------------------------------------------------------------------------
// Wave-level GEMM, one 16-row M tile:
//   acc[nt] (16x16 f32 C-layout) += A(16 x 32*KT bf16, LDS row-major, stride Kp)
//                                    x B fragments (global, frag = nt*KT+kt,
//                                      32 lanes x 16 bf16 each)
// A frag per lane: row = lane&15, K = kt*32 + (lane>>4)*8 + {0..7, 16..23}.
// ---------------------------------------------------------------------------
template <int NT, int KT>
__device__ __forceinline__ void mm(f32x8* acc, const bf16* A, int Kp,
                                   const bf16* __restrict__ B) {
  const int lane = threadIdx.x & 31;
  const int row = lane & 15;
  const int kb = (lane >> 4) * 8;
#pragma unroll
  for (int kt = 0; kt < KT; ++kt) {
    const bf16* pa = A + row * Kp + kt * 32 + kb;
    bf16x8 lo = *(const bf16x8*)(pa);
    bf16x8 hi = *(const bf16x8*)(pa + 16);
    bf16x16 a;
#pragma unroll
    for (int j = 0; j < 8; ++j) { a[j] = lo[j]; a[j + 8] = hi[j]; }
#pragma unroll
    for (int nt = 0; nt < NT; ++nt) {
      bf16x16 b = *(const bf16x16*)(B + (nt * KT + kt) * 512 + lane * 16);
      acc[nt] = wmma_bf16(a, b, acc[nt]);
    }
  }
}

// Two M tiles (32 rows) sharing each B-fragment load: halves weight traffic.
template <int NT, int KT>
__device__ __forceinline__ void mm2(f32x8 (*acc)[NT], const bf16* A, int Kp,
                                    const bf16* __restrict__ B) {
  const int lane = threadIdx.x & 31;
  const int row = lane & 15;
  const int kb = (lane >> 4) * 8;
#pragma unroll
  for (int kt = 0; kt < KT; ++kt) {
    const bf16* pa0 = A + row * Kp + kt * 32 + kb;
    const bf16* pa1 = pa0 + 16 * Kp;
    bf16x8 l0 = *(const bf16x8*)(pa0), h0 = *(const bf16x8*)(pa0 + 16);
    bf16x8 l1 = *(const bf16x8*)(pa1), h1 = *(const bf16x8*)(pa1 + 16);
    bf16x16 a0, a1;
#pragma unroll
    for (int j = 0; j < 8; ++j) {
      a0[j] = l0[j]; a0[j + 8] = h0[j];
      a1[j] = l1[j]; a1[j + 8] = h1[j];
    }
#pragma unroll
    for (int nt = 0; nt < NT; ++nt) {
      bf16x16 b = *(const bf16x16*)(B + (nt * KT + kt) * 512 + lane * 16);
      acc[0][nt] = wmma_bf16(a0, b, acc[0][nt]);
      acc[1][nt] = wmma_bf16(a1, b, acc[1][nt]);
    }
  }
}

// Scale f32 accumulators by the per-row scalar a_s (row m = mhi + i [+16*t]):
// implements (X*a_s)@W^T == (X@W^T)*a_s without touching the bf16 A path.
template <int NT>
__device__ __forceinline__ void scale1(f32x8* acc, const float* sAs) {
  const int mhi = ((threadIdx.x & 31) >> 4) * 8;
#pragma unroll
  for (int nt = 0; nt < NT; ++nt)
#pragma unroll
    for (int i = 0; i < 8; ++i) acc[nt][i] *= sAs[mhi + i];
}

template <int NT>
__device__ __forceinline__ void scale2(f32x8 (*acc)[NT], const float* sAs) {
  const int mhi = ((threadIdx.x & 31) >> 4) * 8;
#pragma unroll
  for (int t = 0; t < 2; ++t)
#pragma unroll
    for (int nt = 0; nt < NT; ++nt)
#pragma unroll
      for (int i = 0; i < 8; ++i) acc[t][nt][i] *= sAs[t * 16 + mhi + i];
}

// Gate epilogue for the 16-row kernels (upd1/pre1; c0_out=64, c1_out=64):
// writes silu(s) -> tXs, gated v -> tXv*, dv(v) -> tDv; sigmoid(g) left in accS[4..7].
__device__ __forceinline__ void gate_to_lds(
    f32x8* accS, const f32x8* accP, const f32x8 (*accQ)[4],
    const float* __restrict__ b,
    bf16* tXs, bf16* tDv, bf16* tXv0, bf16* tXv1, bf16* tXv2,
    const float* sAs, const float (*sAv)[3]) {
  const int lane = threadIdx.x & 31;
  const int nlo = lane & 15, mhi = (lane >> 4) * 8;
#pragma unroll
  for (int nt = 0; nt < 8; ++nt) {
    int n = nt * 16 + nlo;
    float bb = b[n];
#pragma unroll
    for (int i = 0; i < 8; ++i) {
      float v = accS[nt][i] + bb;
      if (nt < 4) tXs[(mhi + i) * 64 + n] = (bf16)(v * sigm(v));
      else        accS[nt][i] = sigm(v);
    }
  }
#pragma unroll
  for (int nt = 0; nt < 4; ++nt) {
    int o = nt * 16 + nlo;
#pragma unroll
    for (int i = 0; i < 8; ++i) {
      int m = mhi + i;
      float as = sAs[m], sg = accS[4 + nt][i], p = accP[nt][i];
      float a0 = sAv[m][0], a1 = sAv[m][1], a2 = sAv[m][2];
      float v0 = (p * a0 + accQ[0][nt][i] * as) * sg;
      float v1 = (p * a1 + accQ[1][nt][i] * as) * sg;
      float v2 = (p * a2 + accQ[2][nt][i] * as) * sg;
      tXv0[m * 64 + o] = (bf16)v0;
      tXv1[m * 64 + o] = (bf16)v1;
      tXv2[m * 64 + o] = (bf16)v2;
      tDv[m * 64 + o]  = (bf16)(v0 * a0 + v1 * a1 + v2 * a2);
    }
  }
}

// ---------------------------------------------------------------------------
// Weight -> bf16 B-fragment conversion (norm folded, K zero-padded)
// ---------------------------------------------------------------------------
__global__ void k_prep(const float* __restrict__ W, bf16* __restrict__ out,
                       int Nout, int Kreal, int Kp, float scale) {
  int f = blockIdx.x;
  int Ktiles = Kp >> 5;
  int nt = f / Ktiles, kt = f - nt * Ktiles;
  int lane = threadIdx.x;
  int n  = nt * 16 + (lane & 15);
  int kb = kt * 32 + (lane >> 4) * 8;
  bf16* o = out + f * 512 + lane * 16;
#pragma unroll
  for (int j = 0; j < 16; ++j) {
    int k = kb + (j < 8 ? j : j + 8);
    float v = (k < Kreal) ? W[n * Kreal + k] * scale : 0.f;
    o[j] = (bf16)v;
  }
}

// ---------------------------------------------------------------------------
// Embedding TP (c0_in=c1_in=1): elementwise
// ---------------------------------------------------------------------------
__global__ void k_emb(const float* __restrict__ x_s, const float* __restrict__ x_v,
                      const float* __restrict__ na_s, const float* __restrict__ na_v,
                      const float* __restrict__ Wss, const float* __restrict__ Wvs,
                      const float* __restrict__ bb,  const float* __restrict__ Wsv,
                      const float* __restrict__ Wvv,
                      float* __restrict__ h_s, float* __restrict__ h_v) {
  int i = blockIdx.x * blockDim.x + threadIdx.x;
  if (i >= NNODES * 64) return;
  int n = i >> 6, o = i & 63;
  const float nrm = 0.70710678118654752f;  // 1/sqrt(2)
  float xs = x_s[n], as = na_s[n];
  float xv0 = x_v[n * 3], xv1 = x_v[n * 3 + 1], xv2 = x_v[n * 3 + 2];
  float av0 = na_v[n * 3], av1 = na_v[n * 3 + 1], av2 = na_v[n * 3 + 2];
  float dv = xv0 * av0 + xv1 * av1 + xv2 * av2;
  h_s[i] = (xs * as * Wss[o] + dv * Wvs[o]) * nrm + bb[o];
  float sv = xs * Wsv[o] * nrm;
  float vv = Wvv[o] * as * nrm;
  h_v[i * 3 + 0] = sv * av0 + xv0 * vv;
  h_v[i * 3 + 1] = sv * av1 + xv1 * vv;
  h_v[i * 3 + 2] = sv * av2 + xv2 * vv;
}

// ---------------------------------------------------------------------------
// Fused msg1 + msg2 + segment-sum scatter. One wave = 32 edges (2 M tiles),
// phase-split (S -> P -> Q0..Q2) so sigmoid(g)/P stay lane-local in regs; dv accumulates lane-local in LDS f32.
// LDS arena with phase-dead overlays (60 KB < 64 KB static limit).
// ---------------------------------------------------------------------------
__global__ void __launch_bounds__(32) k_msg(
    const float* __restrict__ h_s, const float* __restrict__ h_v,
    const int* __restrict__ src, const int* __restrict__ dst,
    const float* __restrict__ ea_s, const float* __restrict__ ea_v,
    const float* __restrict__ amf,
    const float* __restrict__ b1, const float* __restrict__ b2,
    const bf16* __restrict__ f1ss, const bf16* __restrict__ f1vs,
    const bf16* __restrict__ f1sv, const bf16* __restrict__ f1vv,
    const bf16* __restrict__ f2ss, const bf16* __restrict__ f2vs,
    const bf16* __restrict__ f2sv, const bf16* __restrict__ f2vv,
    float* __restrict__ agg_s, float* __restrict__ agg_v) {
  __shared__ alignas(16) char smem[60416];
  bf16*  aXs = (bf16*)(smem);              // [32][160] 10240 B (dead after P)
  bf16*  aDv = (bf16*)(smem + 10240);      // [32][128] 8192 B (dead after S)
  bf16*  aXv = (bf16*)(smem + 18432);      // 3x[32][128] 24576 B
  bf16*  tXs = (bf16*)(smem + 43008);      // [32][64] 4096 B (msg2 A)
  bf16*  tXv = (bf16*)(smem + 47104);      // 3x[32][64] 12288 B (msg2 A)
  bf16*  tDv = (bf16*)(smem);              // [32][64] 4096 B, overlays aXs
  float* dvf = (float*)(smem + 10240);     // [32][64] f32 8192 B, overlays aDv
  float* sAs = (float*)(smem + 59392);     // [32]
  float* sAv = (float*)(smem + 59520);     // [32][3]
  int*   sSrc = (int*)(smem + 59904);      // [32]
  int*   sDst = (int*)(smem + 60032);      // [32]

  const int lane = threadIdx.x;
  const int e0 = blockIdx.x * 32;
  {
    int e = e0 + lane;
    sSrc[lane] = src[e];
    sDst[lane] = dst[e];
    sAs[lane] = ea_s[e];
    sAv[lane * 3 + 0] = ea_v[e * 3 + 0];
    sAv[lane * 3 + 1] = ea_v[e * 3 + 1];
    sAv[lane * 3 + 2] = ea_v[e * 3 + 2];
  }
  __syncthreads();
  for (int r = 0; r < 32; ++r) {
    int nd = sDst[r], ns = sSrc[r];
    for (int c = lane; c < 160; c += 32) {
      float v = 0.f;
      if (c < 64)       v = h_s[nd * 64 + c];
      else if (c < 128) v = h_s[ns * 64 + (c - 64)];
      else if (c < 130) v = amf[(e0 + r) * 2 + (c - 128)];
      aXs[r * 160 + c] = (bf16)v;
    }
    float a0 = sAv[r * 3], a1 = sAv[r * 3 + 1], a2 = sAv[r * 3 + 2];
    for (int c = lane; c < 128; c += 32) {
      int base = ((c < 64) ? (nd * 64 + c) : (ns * 64 + (c - 64))) * 3;
      float v0 = h_v[base], v1 = h_v[base + 1], v2 = h_v[base + 2];
      aXv[0 * 4096 + r * 128 + c] = (bf16)v0;
      aXv[1 * 4096 + r * 128 + c] = (bf16)v1;
      aXv[2 * 4096 + r * 128 + c] = (bf16)v2;
      aDv[r * 128 + c] = (bf16)(v0 * a0 + v1 * a1 + v2 * a2);
    }
  }
  __syncthreads();

  const int nlo = lane & 15, mhi = (lane >> 4) * 8;

  // ---- msg1 S phase: s = (mi_s @ Wss^T)*a_s + dv @ Wvs^T (+b in epilogue)
  f32x8 accS[2][8];
#pragma unroll
  for (int t = 0; t < 2; ++t)
#pragma unroll
    for (int i = 0; i < 8; ++i) accS[t][i] = fzero();
  mm2<8, 5>(accS, aXs, 160, f1ss);
  scale2<8>(accS, sAs);
  mm2<8, 4>(accS, aDv, 128, f1vs);
#pragma unroll
  for (int t = 0; t < 2; ++t)
#pragma unroll
    for (int nt = 0; nt < 8; ++nt) {
      int n = nt * 16 + nlo;
      float bb = b1[n];
#pragma unroll
      for (int i = 0; i < 8; ++i) {
        float v = accS[t][nt][i] + bb;
        if (nt < 4) tXs[(t * 16 + mhi + i) * 64 + n] = (bf16)(v * sigm(v));
        else        accS[t][nt][i] = sigm(v);  // sg kept lane-local in regs
      }
    }
  __syncthreads();

  // ---- msg1 P phase: P = mi_s @ Wsv^T (kept in regs; consumed lane-local)
  f32x8 accP[2][4];
#pragma unroll
  for (int t = 0; t < 2; ++t)
#pragma unroll
    for (int i = 0; i < 4; ++i) accP[t][i] = fzero();
  mm2<4, 5>(accP, aXs, 160, f1sv);
  __syncthreads();  // aXs dead; tDv overlay becomes legal

  // ---- msg1 Q phases: Q_x = xv_x @ Wvv^T; gated v -> tXv, dv accum in LDS f32
  f32x8 accQ[2][4];
#pragma unroll
  for (int x = 0; x < 3; ++x) {
#pragma unroll
    for (int t = 0; t < 2; ++t)
#pragma unroll
      for (int i = 0; i < 4; ++i) accQ[t][i] = fzero();
    mm2<4, 4>(accQ, aXv + x * 4096, 128, f1vv);
#pragma unroll
    for (int t = 0; t < 2; ++t)
#pragma unroll
      for (int nt = 0; nt < 4; ++nt) {
        int o = nt * 16 + nlo;
#pragma unroll
        for (int i = 0; i < 8; ++i) {
          int m = t * 16 + mhi + i;
          float as = sAs[m], av = sAv[m * 3 + x];
          float vx = (accP[t][nt][i] * av + accQ[t][nt][i] * as) * accS[t][4 + nt][i];
          tXv[x * 2048 + m * 64 + o] = (bf16)vx;
          float prev = (x == 0) ? 0.f : dvf[m * 64 + o];  // lane-local RMW
          dvf[m * 64 + o] = prev + vx * av;
        }
      }
  }
  __syncthreads();
#pragma unroll
  for (int t = 0; t < 2; ++t)
#pragma unroll
    for (int nt = 0; nt < 4; ++nt) {
      int o = nt * 16 + nlo;
#pragma unroll
      for (int i = 0; i < 8; ++i) {
        int m = t * 16 + mhi + i;
        tDv[m * 64 + o] = (bf16)dvf[m * 64 + o];
      }
    }
  __syncthreads();

  // ---- msg2 S phase
#pragma unroll
  for (int t = 0; t < 2; ++t)
#pragma unroll
    for (int i = 0; i < 8; ++i) accS[t][i] = fzero();
  mm2<8, 2>(accS, tXs, 64, f2ss);
  scale2<8>(accS, sAs);
  mm2<8, 2>(accS, tDv, 64, f2vs);
#pragma unroll
  for (int t = 0; t < 2; ++t)
#pragma unroll
    for (int nt = 0; nt < 8; ++nt) {
      int n = nt * 16 + nlo;
      float bb = b2[n];
#pragma unroll
      for (int i = 0; i < 8; ++i) {
        float v = accS[t][nt][i] + bb;
        if (nt < 4) atomicAdd(&agg_s[sDst[t * 16 + mhi + i] * 64 + n], v * sigm(v));
        else        accS[t][nt][i] = sigm(v);
      }
    }

  // ---- msg2 P phase
#pragma unroll
  for (int t = 0; t < 2; ++t)
#pragma unroll
    for (int i = 0; i < 4; ++i) accP[t][i] = fzero();
  mm2<4, 2>(accP, tXs, 64, f2sv);

  // ---- msg2 Q phases + atomic scatter of gated v
#pragma unroll
  for (int x = 0; x < 3; ++x) {
#pragma unroll
    for (int t = 0; t < 2; ++t)
#pragma unroll
      for (int i = 0; i < 4; ++i) accQ[t][i] = fzero();
    mm2<4, 2>(accQ, tXv + x * 2048, 64, f2vv);
#pragma unroll
    for (int t = 0; t < 2; ++t)
#pragma unroll
      for (int nt = 0; nt < 4; ++nt) {
        int o = nt * 16 + nlo;
#pragma unroll
        for (int i = 0; i < 8; ++i) {
          int m = t * 16 + mhi + i;
          float as = sAs[m], av = sAv[m * 3 + x];
          float vx = (accP[t][nt][i] * av + accQ[t][nt][i] * as) * accS[t][4 + nt][i];
          atomicAdd(&agg_v[(sDst[m] * 64 + o) * 3 + x], vx);
        }
      }
  }
}

// ---------------------------------------------------------------------------
// Fused upd1 + upd2 + residual. One wave = 16 nodes (contiguous rows).
// ---------------------------------------------------------------------------
__global__ void __launch_bounds__(32) k_upd(
    float* __restrict__ h_s, float* __restrict__ h_v,
    const float* __restrict__ agg_s, const float* __restrict__ agg_v,
    const float* __restrict__ na_s, const float* __restrict__ na_v,
    const float* __restrict__ b1, const float* __restrict__ b2,
    const bf16* __restrict__ f1ss, const bf16* __restrict__ f1vs,
    const bf16* __restrict__ f1sv, const bf16* __restrict__ f1vv,
    const bf16* __restrict__ f2ss, const bf16* __restrict__ f2vs,
    const bf16* __restrict__ f2sv, const bf16* __restrict__ f2vv) {
  __shared__ alignas(16) bf16 aXs[16 * 128];
  __shared__ alignas(16) bf16 aDv[16 * 128];
  __shared__ alignas(16) bf16 aXv[3][16 * 128];
  __shared__ alignas(16) bf16 tXs[16 * 64];
  __shared__ alignas(16) bf16 tDv[16 * 64];
  __shared__ alignas(16) bf16 tXv[3][16 * 64];
  __shared__ float sAs[16];
  __shared__ float sAv[16][3];

  const int lane = threadIdx.x;
  const int n0 = blockIdx.x * 16;
  if (lane < 16) {
    int n = n0 + lane;
    sAs[lane] = na_s[n];
    sAv[lane][0] = na_v[n * 3 + 0];
    sAv[lane][1] = na_v[n * 3 + 1];
    sAv[lane][2] = na_v[n * 3 + 2];
  }
  __syncthreads();
  for (int r = 0; r < 16; ++r) {
    int n = n0 + r;
    for (int c = lane; c < 128; c += 32) {
      float v = (c < 64) ? h_s[n * 64 + c] : agg_s[n * 64 + (c - 64)];
      aXs[r * 128 + c] = (bf16)v;
    }
    float a0 = sAv[r][0], a1 = sAv[r][1], a2 = sAv[r][2];
    for (int c = lane; c < 128; c += 32) {
      const float* pv = (c < 64) ? &h_v[(n * 64 + c) * 3] : &agg_v[(n * 64 + (c - 64)) * 3];
      float v0 = pv[0], v1 = pv[1], v2 = pv[2];
      aXv[0][r * 128 + c] = (bf16)v0;
      aXv[1][r * 128 + c] = (bf16)v1;
      aXv[2][r * 128 + c] = (bf16)v2;
      aDv[r * 128 + c] = (bf16)(v0 * a0 + v1 * a1 + v2 * a2);
    }
  }
  __syncthreads();

  f32x8 accS[8], accP[4], accQ[3][4];
#pragma unroll
  for (int i = 0; i < 8; ++i) accS[i] = fzero();
#pragma unroll
  for (int i = 0; i < 4; ++i) {
    accP[i] = fzero(); accQ[0][i] = fzero(); accQ[1][i] = fzero(); accQ[2][i] = fzero();
  }
  mm<8, 4>(accS, aXs, 128, f1ss);
  scale1<8>(accS, sAs);
  mm<8, 4>(accS, aDv, 128, f1vs);
  mm<4, 4>(accP, aXs, 128, f1sv);
  mm<4, 4>(accQ[0], aXv[0], 128, f1vv);
  mm<4, 4>(accQ[1], aXv[1], 128, f1vv);
  mm<4, 4>(accQ[2], aXv[2], 128, f1vv);
  gate_to_lds(accS, accP, accQ, b1, tXs, tDv, tXv[0], tXv[1], tXv[2], sAs, sAv);
  __syncthreads();

#pragma unroll
  for (int i = 0; i < 4; ++i) {
    accS[i] = fzero(); accP[i] = fzero();
    accQ[0][i] = fzero(); accQ[1][i] = fzero(); accQ[2][i] = fzero();
  }
  // upd2 (no gate, s_tot = 64)
  mm<4, 2>(accS, tXs, 64, f2ss);
  scale1<4>(accS, sAs);
  mm<4, 2>(accS, tDv, 64, f2vs);
  mm<4, 2>(accP, tXs, 64, f2sv);
  mm<4, 2>(accQ[0], tXv[0], 64, f2vv);
  mm<4, 2>(accQ[1], tXv[1], 64, f2vv);
  mm<4, 2>(accQ[2], tXv[2], 64, f2vv);

  const int nlo = lane & 15, mhi = (lane >> 4) * 8;
#pragma unroll
  for (int nt = 0; nt < 4; ++nt) {
    int n = nt * 16 + nlo;
    float bb = b2[n];
#pragma unroll
    for (int i = 0; i < 8; ++i) {
      int m = mhi + i;
      int node = n0 + m;
      h_s[node * 64 + n] += accS[nt][i] + bb;
      float as = sAs[m], p = accP[nt][i];
      int vb = (node * 64 + n) * 3;
      h_v[vb + 0] += p * sAv[m][0] + accQ[0][nt][i] * as;
      h_v[vb + 1] += p * sAv[m][1] + accQ[1][nt][i] * as;
      h_v[vb + 2] += p * sAv[m][2] + accQ[2][nt][i] * as;
    }
  }
}

// ---------------------------------------------------------------------------
// Fused pre1 + pre2 + pooled (atomic) graph sum. One wave = 16 nodes.
// ---------------------------------------------------------------------------
__global__ void __launch_bounds__(32) k_pre(
    const float* __restrict__ h_s, const float* __restrict__ h_v,
    const float* __restrict__ na_s, const float* __restrict__ na_v,
    const float* __restrict__ b1, const float* __restrict__ b2,
    const bf16* __restrict__ f1ss, const bf16* __restrict__ f1vs,
    const bf16* __restrict__ f1sv, const bf16* __restrict__ f1vv,
    const bf16* __restrict__ f2ss, const bf16* __restrict__ f2vs,
    const int* __restrict__ batch, float* __restrict__ sums) {
  __shared__ alignas(16) bf16 aXs[16 * 64];
  __shared__ alignas(16) bf16 aDv[16 * 64];
  __shared__ alignas(16) bf16 aXv[3][16 * 64];
  __shared__ alignas(16) bf16 tXs[16 * 64];
  __shared__ alignas(16) bf16 tDv[16 * 64];
  __shared__ alignas(16) bf16 tXv[3][16 * 64];
  __shared__ float sAs[16];
  __shared__ float sAv[16][3];
  __shared__ int sB[16];

  const int lane = threadIdx.x;
  const int n0 = blockIdx.x * 16;
  if (lane < 16) {
    int n = n0 + lane;
    sAs[lane] = na_s[n];
    sAv[lane][0] = na_v[n * 3 + 0];
    sAv[lane][1] = na_v[n * 3 + 1];
    sAv[lane][2] = na_v[n * 3 + 2];
    sB[lane] = batch[n];
  }
  __syncthreads();
  for (int r = 0; r < 16; ++r) {
    int n = n0 + r;
    for (int c = lane; c < 64; c += 32) aXs[r * 64 + c] = (bf16)h_s[n * 64 + c];
    float a0 = sAv[r][0], a1 = sAv[r][1], a2 = sAv[r][2];
    for (int c = lane; c < 64; c += 32) {
      const float* pv = &h_v[(n * 64 + c) * 3];
      float v0 = pv[0], v1 = pv[1], v2 = pv[2];
      aXv[0][r * 64 + c] = (bf16)v0;
      aXv[1][r * 64 + c] = (bf16)v1;
      aXv[2][r * 64 + c] = (bf16)v2;
      aDv[r * 64 + c] = (bf16)(v0 * a0 + v1 * a1 + v2 * a2);
    }
  }
  __syncthreads();

  f32x8 accS[8], accP[4], accQ[3][4];
#pragma unroll
  for (int i = 0; i < 8; ++i) accS[i] = fzero();
#pragma unroll
  for (int i = 0; i < 4; ++i) {
    accP[i] = fzero(); accQ[0][i] = fzero(); accQ[1][i] = fzero(); accQ[2][i] = fzero();
  }
  mm<8, 2>(accS, aXs, 64, f1ss);
  scale1<8>(accS, sAs);
  mm<8, 2>(accS, aDv, 64, f1vs);
  mm<4, 2>(accP, aXs, 64, f1sv);
  mm<4, 2>(accQ[0], aXv[0], 64, f1vv);
  mm<4, 2>(accQ[1], aXv[1], 64, f1vv);
  mm<4, 2>(accQ[2], aXv[2], 64, f1vv);
  gate_to_lds(accS, accP, accQ, b1, tXs, tDv, tXv[0], tXv[1], tXv[2], sAs, sAv);
  __syncthreads();

#pragma unroll
  for (int i = 0; i < 8; ++i) accS[i] = fzero();
  // pre2: s only (c1_out = 0, s_tot = 128)
  mm<8, 2>(accS, tXs, 64, f2ss);
  scale1<8>(accS, sAs);
  mm<8, 2>(accS, tDv, 64, f2vs);

  const int nlo = lane & 15, mhi = (lane >> 4) * 8;
#pragma unroll
  for (int nt = 0; nt < 8; ++nt) {
    int n = nt * 16 + nlo;
    float bb = b2[n];
#pragma unroll
    for (int i = 0; i < 8; ++i) {
      int m = mhi + i;
      atomicAdd(&sums[sB[m] * 128 + n], accS[nt][i] + bb);
    }
  }
}

__global__ void k_count(const int* __restrict__ batch, float* __restrict__ cnt, int n) {
  int i = blockIdx.x * blockDim.x + threadIdx.x;
  if (i < n) atomicAdd(&cnt[batch[i]], 1.0f);
}

// Pooled mean + 2-layer MLP head. 1 block of 128 threads.
__global__ void k_final(const float* __restrict__ sums, const float* __restrict__ cnt,
                        const float* __restrict__ W1, const float* __restrict__ b1,
                        const float* __restrict__ W2, const float* __restrict__ b2,
                        float* __restrict__ out) {
  __shared__ float g[128];
  __shared__ float red[128];
  const int t = threadIdx.x;
  const float nrm = 0.08838834764831845f;  // 1/sqrt(128)
  for (int gi = 0; gi < NGRAPH; ++gi) {
    float c = fmaxf(cnt[gi], 1.0f);
    g[t] = sums[gi * 128 + t] / c;
    __syncthreads();
    float acc = 0.f;
    for (int k = 0; k < 128; ++k) acc += g[k] * W1[t * 128 + k];
    float z = acc * nrm + b1[t];
    z = z * sigm(z);
    red[t] = z * W2[t];
    __syncthreads();
    for (int s = 64; s > 0; s >>= 1) {
      if (t < s) red[t] += red[t + s];
      __syncthreads();
    }
    if (t == 0) out[gi] = red[0] * nrm + b2[0];
    __syncthreads();
  }
}

// ---------------------------------------------------------------------------
extern "C" void kernel_launch(void* const* d_in, const int* in_sizes, int n_in,
                              void* d_out, int out_size, void* d_ws, size_t ws_size,
                              hipStream_t stream) {
  (void)in_sizes; (void)n_in; (void)out_size; (void)ws_size;
  const float* x_s  = (const float*)d_in[0];
  const float* x_v  = (const float*)d_in[1];
  const int*   eidx = (const int*)d_in[2];
  const float* ea_s = (const float*)d_in[3];
  const float* ea_v = (const float*)d_in[4];
  const float* na_s = (const float*)d_in[5];
  const float* na_v = (const float*)d_in[6];
  const float* amf  = (const float*)d_in[7];
  const int*   batch = (const int*)d_in[8];
  const int* src = eidx;
  const int* dst = eidx + NEDGES;

  char* ws = (char*)d_ws;
  size_t off = 0;
  auto alloc = [&](size_t bytes) -> void* {
    void* p = ws + off;
    off = (off + bytes + 255) & ~(size_t)255;
    return p;
  };
  float* h_s   = (float*)alloc((size_t)NNODES * 64 * 4);
  float* h_v   = (float*)alloc((size_t)NNODES * 64 * 3 * 4);
  float* agg_s = (float*)alloc((size_t)NNODES * 64 * 4);
  float* agg_v = (float*)alloc((size_t)NNODES * 64 * 3 * 4);
  float* sums  = (float*)alloc((size_t)NGRAPH * 128 * 4);
  float* cnt   = (float*)alloc((size_t)NGRAPH * 4);

  // bf16 fragment sets (norm folded). Param leaves assumed flattened in
  // insertion order: 9..13 emb{W_ss,W_vs,b,W_sv,W_vv}, then per layer
  // msg1,msg2,upd1,upd2 (5 leaves each), pre1 @54, pre2 @59, post @64..67.
  struct FD { int idx, nout, kreal, kp; float sc; };
  const float n258 = 0.06225728063f;  // 1/sqrt(258)
  const float n128 = 0.08838834765f;  // 1/sqrt(128)
  const float n256 = 0.0625f;         // 1/sqrt(256)
  FD fds[38];
  int nf = 0;
  for (int li = 0; li < 2; ++li) {
    int b = 14 + 20 * li;
    FD l[16] = {
      {b + 0, 128, 130, 160, n258}, {b + 1, 128, 128, 128, n258},
      {b + 3,  64, 130, 160, n258}, {b + 4,  64, 128, 128, n258},
      {b + 5, 128,  64,  64, n128}, {b + 6, 128,  64,  64, n128},
      {b + 8,  64,  64,  64, n128}, {b + 9,  64,  64,  64, n128},
      {b + 10, 128, 128, 128, n256}, {b + 11, 128, 128, 128, n256},
      {b + 13,  64, 128, 128, n256}, {b + 14,  64, 128, 128, n256},
      {b + 15,  64,  64,  64, n128}, {b + 16,  64,  64,  64, n128},
      {b + 18,  64,  64,  64, n128}, {b + 19,  64,  64,  64, n128}};
    for (int i = 0; i < 16; ++i) fds[nf++] = l[i];
  }
  FD tail[6] = {{54, 128, 64, 64, n128}, {55, 128, 64, 64, n128},
                {57,  64, 64, 64, n128}, {58,  64, 64, 64, n128},
                {59, 128, 64, 64, n128}, {60, 128, 64, 64, n128}};
  for (int i = 0; i < 6; ++i) fds[nf++] = tail[i];

  bf16* fp[38];
  for (int i = 0; i < nf; ++i) {
    int frags = (fds[i].kp / 32) * (fds[i].nout / 16);
    fp[i] = (bf16*)alloc((size_t)frags * 512 * sizeof(bf16));
  }
  for (int i = 0; i < nf; ++i) {
    int frags = (fds[i].kp / 32) * (fds[i].nout / 16);
    k_prep<<<frags, 32, 0, stream>>>((const float*)d_in[fds[i].idx], fp[i],
                                     fds[i].nout, fds[i].kreal, fds[i].kp, fds[i].sc);
  }

  k_emb<<<(NNODES * 64 + 255) / 256, 256, 0, stream>>>(
      x_s, x_v, na_s, na_v,
      (const float*)d_in[9], (const float*)d_in[10], (const float*)d_in[11],
      (const float*)d_in[12], (const float*)d_in[13], h_s, h_v);

  for (int li = 0; li < 2; ++li) {
    int pb = 14 + 20 * li;
    bf16** L = &fp[li * 16];
    hipMemsetAsync(agg_s, 0, (size_t)NNODES * 64 * 4, stream);
    hipMemsetAsync(agg_v, 0, (size_t)NNODES * 64 * 3 * 4, stream);
    k_msg<<<NEDGES / 32, 32, 0, stream>>>(
        h_s, h_v, src, dst, ea_s, ea_v, amf,
        (const float*)d_in[pb + 2], (const float*)d_in[pb + 7],
        L[0], L[1], L[2], L[3], L[4], L[5], L[6], L[7], agg_s, agg_v);
    k_upd<<<NNODES / 16, 32, 0, stream>>>(
        h_s, h_v, agg_s, agg_v, na_s, na_v,
        (const float*)d_in[pb + 12], (const float*)d_in[pb + 17],
        L[8], L[9], L[10], L[11], L[12], L[13], L[14], L[15]);
  }

  hipMemsetAsync(sums, 0, (size_t)NGRAPH * 128 * 4, stream);
  hipMemsetAsync(cnt, 0, (size_t)NGRAPH * 4, stream);
  k_pre<<<NNODES / 16, 32, 0, stream>>>(
      h_s, h_v, na_s, na_v,
      (const float*)d_in[56], (const float*)d_in[61],
      fp[32], fp[33], fp[34], fp[35], fp[36], fp[37], batch, sums);
  k_count<<<(NNODES + 255) / 256, 256, 0, stream>>>(batch, cnt, NNODES);
  k_final<<<1, 128, 0, stream>>>(sums, cnt,
      (const float*)d_in[64], (const float*)d_in[65],
      (const float*)d_in[66], (const float*)d_in[67], (float*)d_out);
}
